// Context_Win_Graph_72353019069082
// MI455X (gfx1250) — matrix operand order repaired and checked by reading, not compiled, not run on hardware
//
#include <hip/hip_runtime.h>
#include <hip/hip_bf16.h>

// ---------------------------------------------------------------------------
// CDNA5 (gfx1250) fused windowed graph-attention pipeline, bf16 WMMA GEMMs
// + async global->LDS staging (ASYNCcnt) via gfx1250 builtins.
// ---------------------------------------------------------------------------

typedef __attribute__((ext_vector_type(16))) __bf16 v16bf;
typedef __attribute__((ext_vector_type(8)))  __bf16 v8bf;
typedef __attribute__((ext_vector_type(8)))  float  v8f;
typedef int vec4i __attribute__((vector_size(16)));   // matches builtin param type

#define WMMA_BF16(a, b, c) \
  __builtin_amdgcn_wmma_f32_16x16x32_bf16(false, (a), false, (b), (short)0, (c), false, false)

#define AS1 __attribute__((address_space(1)))
#define AS3 __attribute__((address_space(3)))

#if defined(__has_builtin)
#if __has_builtin(__builtin_amdgcn_global_load_async_to_lds_b128)
#define HAVE_ASYNC_LDS 1
#endif
#if __has_builtin(__builtin_amdgcn_global_store_async_from_lds_b128)
#define HAVE_ASYNC_ST 1
#endif
#endif
#ifndef HAVE_ASYNC_LDS
#define HAVE_ASYNC_LDS 0
#endif
#ifndef HAVE_ASYNC_ST
#define HAVE_ASYNC_ST 0
#endif

static constexpr int B_  = 32;
static constexpr int H_  = 4;
static constexpr int L_  = 2048;
static constexpr int D_  = 64;
static constexpr int WN  = 63;   // win_num = P - win + 1
static constexpr int I_  = 64;   // N*win rows per window

static constexpr long TOT_WX   = (long)B_ * H_ * WN * I_ * D_;   // 33,030,144
static constexpr long TOT_OUT  = (long)B_ * L_ * H_ * D_;        // 16,777,216
static constexpr long TOT_STAT = (long)B_ * H_ * 64 * 64;        // 524,288

// workspace layout (bytes)
static constexpr size_t OFF_WXA  = 0;                                  // bf16 [b][h][w][i][d]
static constexpr size_t SZ_WXA   = (size_t)TOT_WX * 2;
static constexpr size_t OFF_VT   = OFF_WXA + SZ_WXA;                   // bf16 [b][w][c][j]
static constexpr size_t SZ_VT    = (size_t)B_ * WN * 256 * 64 * 2;
static constexpr size_t OFF_OUTB = OFF_VT + SZ_VT;                     // f32 [b][h][w][i][d]
static constexpr size_t SZ_OUTB  = (size_t)TOT_WX * 4;
static constexpr size_t OFF_W1T  = OFF_OUTB + SZ_OUTB;                 // bf16 [n][k] 64x64
static constexpr size_t OFF_W2T  = OFF_W1T + 64 * 64 * 2;
static constexpr size_t OFF_WVT  = OFF_W2T + 64 * 64 * 2;              // bf16 [n][k] 256x256
static constexpr size_t OFF_GSUM = OFF_WVT + 256 * 256 * 2;            // f32 [b][h][i][j]
static constexpr size_t OFF_GSQ  = OFF_GSUM + (size_t)TOT_STAT * 4;
// total ~268.6 MB

// ---------------------------------------------------------------------------
// async global<->LDS helpers (fallback: VGPR round-trip copies)
// ---------------------------------------------------------------------------
__device__ __forceinline__ void cp16(void* dst, const void* src, int bytes, int tid, int nthreads) {
  uint4* d = (uint4*)dst;
  const uint4* s = (const uint4*)src;
  int n = bytes >> 4;
  for (int i = tid; i < n; i += nthreads) d[i] = s[i];
}

// one 16B async chunk: global -> LDS, no VGPR data movement (ASYNCcnt)
__device__ __forceinline__ void async_cp16_one(void* lds, const void* gsrc) {
#if HAVE_ASYNC_LDS
  __builtin_amdgcn_global_load_async_to_lds_b128(
      (AS1 vec4i*)(unsigned long long)gsrc,
      (AS3 vec4i*)(unsigned long long)lds, 0, 0);
#else
  *(uint4*)lds = *(const uint4*)gsrc;
#endif
}

__device__ __forceinline__ void async_cp16(void* dst, const void* src, int bytes, int tid, int nthreads) {
  int n = bytes >> 4;
  for (int i = tid; i < n; i += nthreads)
    async_cp16_one((char*)dst + (size_t)i * 16, (const char*)src + (size_t)i * 16);
}

__device__ __forceinline__ void async_wait0() {
#if HAVE_ASYNC_LDS
#if defined(__has_builtin) && __has_builtin(__builtin_amdgcn_s_wait_asynccnt)
  __builtin_amdgcn_s_wait_asynccnt(0);
#else
  asm volatile("s_wait_asynccnt 0x0" ::: "memory");
#endif
#endif
}

// one 16B async chunk: LDS -> global
__device__ __forceinline__ void async_st16_one(void* gdst, const void* lds) {
#if HAVE_ASYNC_ST
  __builtin_amdgcn_global_store_async_from_lds_b128(
      (AS1 vec4i*)(unsigned long long)gdst,
      (AS3 vec4i*)(unsigned long long)lds, 0, 0);
#else
  *(uint4*)gdst = *(const uint4*)lds;
#endif
}

// ---------------------------------------------------------------------------
// WMMA fragment helpers (ISA 7.12.2 layouts, wave32)
// ---------------------------------------------------------------------------

// A operand: 16x32 bf16 tile from row-major [M][K] source.
// lane<16: row=lane,    K = k0+{0..7, 16..23}
// lane>=16: row=lane-16, K = k0+{8..15, 24..31}
__device__ __forceinline__ v16bf load_a_frag(const __bf16* src, int ld, int row0, int k0, int lane) {
  int r = lane & 15, hh = lane >> 4;
  const __bf16* p = src + (size_t)(row0 + r) * ld + k0 + hh * 8;
  v8bf lo = *(const v8bf*)p;
  v8bf hi = *(const v8bf*)(p + 16);
  v16bf a;
#pragma unroll
  for (int i = 0; i < 8; ++i) { a[i] = lo[i]; a[i + 8] = hi[i]; }
  return a;
}

// B operand: 32x16 bf16 tile from [N][K] (column-major of the math B) source.
// lane<16: col n=lane, K=k0+{0..7, 16..23} ; lane>=16: col n=lane-16, K=k0+{8..15, 24..31}
__device__ __forceinline__ v16bf load_b_frag(const __bf16* srcT, int ld, int n0, int k0, int lane) {
  int r = lane & 15, hh = lane >> 4;
  const __bf16* p = srcT + (size_t)(n0 + r) * ld + k0 + hh * 8;
  v8bf lo = *(const v8bf*)p;
  v8bf hi = *(const v8bf*)(p + 16);
  v16bf b;
#pragma unroll
  for (int i = 0; i < 8; ++i) { b[i] = lo[i]; b[i + 8] = hi[i]; }
  return b;
}

// C/D: VGPR v holds (M = v + 8*(lane>=16), N = n0 + lane%16)
__device__ __forceinline__ void store_c_bf16(__bf16* dst, int ld, int row0, int col0, int lane, v8f acc) {
  int r = lane & 15, hh = lane >> 4;
#pragma unroll
  for (int v = 0; v < 8; ++v)
    dst[(size_t)(row0 + hh * 8 + v) * ld + col0 + r] = (__bf16)acc[v];
}

__device__ __forceinline__ void store_c_f32(float* dst, int ld, int row0, int col0, int lane, v8f acc) {
  int r = lane & 15, hh = lane >> 4;
#pragma unroll
  for (int v = 0; v < 8; ++v)
    dst[(size_t)(row0 + hh * 8 + v) * ld + col0 + r] = acc[v];
}

// ---------------------------------------------------------------------------
// K0: zero accumulators + loss slot
// ---------------------------------------------------------------------------
__global__ void zero_kernel(float* stats, float* lossp) {
  int idx = blockIdx.x * 256 + threadIdx.x;
  if (idx < 2 * (int)TOT_STAT) stats[idx] = 0.0f;
  if (idx == 0) *lossp = 0.0f;
}

// ---------------------------------------------------------------------------
// K1: transpose weights to bf16 [N][K]
// ---------------------------------------------------------------------------
__global__ void prep_kernel(const float* W1, const float* W2, const float* Wv,
                            __bf16* W1t, __bf16* W2t, __bf16* WvT) {
  int idx = blockIdx.x * 256 + threadIdx.x;
  if (idx < 4096) {
    int n = idx >> 6, k = idx & 63;
    W1t[idx] = (__bf16)W1[k * 64 + n];
  } else if (idx < 8192) {
    int q = idx - 4096;
    int n = q >> 6, k = q & 63;
    W2t[q] = (__bf16)W2[k * 64 + n];
  } else if (idx < 8192 + 65536) {
    int q = idx - 8192;
    int n = q >> 8, k = q & 255;
    WvT[q] = (__bf16)Wv[k * 256 + n];
  }
}

// ---------------------------------------------------------------------------
// K2: materialize the reshape-scrambled wx as bf16 [b][h][w][i][d]
// ---------------------------------------------------------------------------
__global__ void gather_wx_kernel(const float* __restrict__ x, __bf16* __restrict__ wxA) {
  long idx = (long)blockIdx.x * 256 + threadIdx.x;
  if (idx >= TOT_WX) return;
  int d  = (int)(idx & 63);
  int i  = (int)((idx >> 6) & 63);
  long r0 = idx >> 12;
  int w  = (int)(r0 % WN);
  long r1 = r0 / WN;
  int h  = (int)(r1 & 3);
  int b  = (int)(r1 >> 2);
  int n  = i >> 1;
  int f  = (i & 1) * 4032 + w * 64 + d;   // flat index inside one n-block
  int dsrc = f / 126;
  int t    = f - dsrc * 126;
  int wsrc = t >> 1;
  int j    = t & 1;
  int p    = wsrc + j;                    // row within P
  float v = x[(((size_t)(b * H_ + h) * L_) + n * 64 + p) * D_ + dsrc];
  wxA[idx] = (__bf16)v;
}

// ---------------------------------------------------------------------------
// K3: v = wx @ Wv + bv, output transposed vt[b][w][c][j] (B-operand layout)
// grid = B*WN blocks of 256
// ---------------------------------------------------------------------------
__global__ void __launch_bounds__(256) v_kernel(const __bf16* __restrict__ wxA,
                                                const __bf16* __restrict__ WvT,
                                                const float* __restrict__ bv,
                                                __bf16* __restrict__ vt) {
  __shared__ __align__(16) __bf16 sAv[64 * 256];   // 32 KB: rows j, cols c=(h,d)
  int blk = blockIdx.x;           // b*WN + w
  int tid = threadIdx.x;
  int wave = tid >> 5, lane = tid & 31;
  int b = blk / WN, w = blk % WN;

  // assemble 64x256 A tile from 4 h-blocks of wxA (async: global -> LDS direct)
  for (int h = 0; h < 4; ++h) {
    const char* src = (const char*)(wxA + ((size_t)((b * H_ + h) * WN) + w) * (I_ * D_));
    for (int ci = tid; ci < 512; ci += 256) {
      int jrow = ci >> 3;
      int d0 = (ci & 7) * 8;
      async_cp16_one(sAv + (size_t)jrow * 256 + h * 64 + d0, src + (size_t)ci * 16);
    }
  }
  async_wait0();
  __syncthreads();

  __bf16* vtBlk = vt + (size_t)blk * 256 * 64;
  for (int t = 0; t < 8; ++t) {
    int tile = wave * 8 + t;          // 64 tiles: 4 (mt) x 16 (nt)
    int mt = tile >> 4, nt = tile & 15;
    int r = lane & 15, hh = lane >> 4;
    float bb = bv[nt * 16 + r];
    v8f acc = {bb, bb, bb, bb, bb, bb, bb, bb};
    __builtin_prefetch(WvT + (size_t)(nt * 16) * 256, 0, 3);
#pragma unroll
    for (int k0 = 0; k0 < 256; k0 += 32) {
      v16bf a  = load_a_frag(sAv, 256, mt * 16, k0, lane);
      v16bf bf = load_b_frag(WvT, 256, nt * 16, k0, lane);
      acc = WMMA_BF16(a, bf, acc);
    }
    // store transposed: vt[c][j], 8 consecutive j per lane -> one 16B store
    int col = nt * 16 + r;
    int jb = mt * 16 + hh * 8;
    __bf16 tmp[8];
#pragma unroll
    for (int v = 0; v < 8; ++v) tmp[v] = (__bf16)acc[v];
    *(uint4*)(vtBlk + (size_t)col * 64 + jb) = *(uint4*)tmp;
  }
}

// ---------------------------------------------------------------------------
// K4: fused q,kk -> Awin -> topk/tanh/norm -> adj -> out = adj @ v
// one block per (b,h,w); grid = B*H*WN = 8064
// ---------------------------------------------------------------------------
__global__ void __launch_bounds__(256) qk_adj_out_kernel(
    const __bf16* __restrict__ wxA, const __bf16* __restrict__ vt,
    const __bf16* __restrict__ W1t, const __bf16* __restrict__ W2t,
    const float* __restrict__ b1, const float* __restrict__ b2,
    const float* __restrict__ alphap,
    float* __restrict__ gsum, float* __restrict__ gsq,
    float* __restrict__ outbuf) {
  __shared__ __align__(16) unsigned char smem[40 * 1024];
  __bf16* sW1  = (__bf16*)smem;               // 8 KB  (stage A)
  __bf16* sW2  = (__bf16*)(smem + 8192);      // 8 KB  (stage A)
  float*  sS   = (float*)smem;                // 16 KB (overlays W1/W2 later)
  __bf16* sA   = (__bf16*)(smem + 16384);     // 8 KB  wx tile
  __bf16* sV   = (__bf16*)(smem + 16384);     // overlays sA after q/kk
  __bf16* sQ   = (__bf16*)(smem + 24576);     // 8 KB
  __bf16* sK   = (__bf16*)(smem + 32768);     // 8 KB
  __bf16* sAdj = (__bf16*)(smem + 32768);     // overlays sK after Awin

  int blk = blockIdx.x;                       // (b*H + h)*WN + w
  int w = blk % WN;
  int bh = blk / WN;
  int h = bh % H_;
  int b = bh / H_;
  int tid = threadIdx.x;
  int wave = tid >> 5, lane = tid & 31;

  // async stage: wx tile + both weight tiles straight into LDS
  async_cp16(sA, wxA + (size_t)blk * (I_ * D_), I_ * D_ * 2, tid, 256);
  async_cp16(sW1, W1t, 64 * 64 * 2, tid, 256);
  async_cp16(sW2, W2t, 64 * 64 * 2, tid, 256);
  async_wait0();
  __syncthreads();

  // ---- q = wx@W1+b1 ; kk = wx@W2+b2 (each wave: 2 of 16 tiles) ----
#pragma unroll
  for (int t = 0; t < 2; ++t) {
    int tile = wave * 2 + t;
    int mt = tile >> 2, nt = tile & 3;
    float bb1 = b1[nt * 16 + (lane & 15)];
    float bb2 = b2[nt * 16 + (lane & 15)];
    v8f accq = {bb1, bb1, bb1, bb1, bb1, bb1, bb1, bb1};
    v8f acck = {bb2, bb2, bb2, bb2, bb2, bb2, bb2, bb2};
#pragma unroll
    for (int k0 = 0; k0 < 64; k0 += 32) {
      v16bf a   = load_a_frag(sA, 64, mt * 16, k0, lane);
      v16bf bw1 = load_b_frag(sW1, 64, nt * 16, k0, lane);
      v16bf bw2 = load_b_frag(sW2, 64, nt * 16, k0, lane);
      accq = WMMA_BF16(a, bw1, accq);
      acck = WMMA_BF16(a, bw2, acck);
    }
    store_c_bf16(sQ, 64, mt * 16, nt * 16, lane, accq);
    store_c_bf16(sK, 64, mt * 16, nt * 16, lane, acck);
  }
  __syncthreads();

  // ---- stream v tile in async (overlays sA) while Awin = q @ kk^T -> sS ----
  async_cp16((void*)sV, vt + ((size_t)(b * WN + w) * 256 + h * 64) * 64, 64 * 64 * 2, tid, 256);
#pragma unroll
  for (int t = 0; t < 2; ++t) {
    int tile = wave * 2 + t;
    int mt = tile >> 2, nt = tile & 3;
    v8f acc = {};
#pragma unroll
    for (int k0 = 0; k0 < 64; k0 += 32) {
      v16bf a  = load_a_frag(sQ, 64, mt * 16, k0, lane);
      v16bf bb = load_b_frag(sK, 64, nt * 16, k0, lane);   // sK rows j contiguous over d
      acc = WMMA_BF16(a, bb, acc);
    }
    store_c_f32(sS, 64, mt * 16, nt * 16, lane, acc);
  }
  __syncthreads();

  // ---- top-k mask (keep largest |.|), tanh, L1-normalize; adj -> sAdj ----
  float alpha = alphap[0];
  int ksmall = (int)(alpha * 64.0f);    // number of smallest-|.| entries zeroed
  for (int rr = 0; rr < 8; ++rr) {
    int row = wave * 8 + rr;
    int c0 = lane, c1 = lane + 32;
    float e0 = sS[row * 64 + c0];
    float e1 = sS[row * 64 + c1];
    float a0 = fabsf(e0), a1 = fabsf(e1);
    int cnt0 = 0, cnt1 = 0;
    for (int t2 = 0; t2 < 64; ++t2) {
      float u = fabsf(sS[row * 64 + t2]);
      cnt0 += (u < a0) || (u == a0 && t2 < c0);
      cnt1 += (u < a1) || (u == a1 && t2 < c1);
    }
    float t0 = (cnt0 >= ksmall) ? tanhf(e0) : 0.0f;
    float t1 = (cnt1 >= ksmall) ? tanhf(e1) : 0.0f;
    float s = fabsf(t0) + fabsf(t1);
#pragma unroll
    for (int off = 16; off > 0; off >>= 1) s += __shfl_xor(s, off, 32);
    float denom = fmaxf(s, 1e-12f);
    float v0 = t0 / denom, v1 = t1 / denom;
    sAdj[row * 64 + c0] = (__bf16)v0;
    sAdj[row * 64 + c1] = (__bf16)v1;
    int gidx = (bh * 64 + row) * 64;
    atomicAdd(&gsum[gidx + c0], v0);
    atomicAdd(&gsum[gidx + c1], v1);
    atomicAdd(&gsq[gidx + c0], v0 * v0);
    atomicAdd(&gsq[gidx + c1], v1 * v1);
  }
  async_wait0();      // sV must be resident before out GEMM
  __syncthreads();

  // ---- out = adj @ v : A=sAdj [i][j], B=sV [d][j] -> sS fp32 ----
#pragma unroll
  for (int t = 0; t < 2; ++t) {
    int tile = wave * 2 + t;
    int mt = tile >> 2, nt = tile & 3;
    v8f acc = {};
#pragma unroll
    for (int k0 = 0; k0 < 64; k0 += 32) {
      v16bf a  = load_a_frag(sAdj, 64, mt * 16, k0, lane);
      v16bf bb = load_b_frag(sV, 64, nt * 16, k0, lane);
      acc = WMMA_BF16(a, bb, acc);
    }
    store_c_f32(sS, 64, mt * 16, nt * 16, lane, acc);
  }
  __syncthreads();

  // async LDS -> global store of the out tile (waits folded into S_ENDPGM)
  {
    char* gdst = (char*)(outbuf + (size_t)blk * (I_ * D_));
    int n = (I_ * D_ * 4) >> 4;
    for (int i = tid; i < n; i += 256)
      async_st16_one(gdst + (size_t)i * 16, (const char*)sS + (size_t)i * 16);
  }
}

// ---------------------------------------------------------------------------
// K5: reorder + edge-pad + pair-average epilogue -> d_out
// ---------------------------------------------------------------------------
__global__ void merge_kernel(const float* __restrict__ outbuf, float* __restrict__ out) {
  long idx = (long)blockIdx.x * 256 + threadIdx.x;
  if (idx >= TOT_OUT) return;
  int d = (int)(idx & 63);
  int h = (int)((idx >> 6) & 3);
  int p = (int)((idx >> 8) & 63);
  int n = (int)((idx >> 14) & 31);
  int b = (int)(idx >> 19);
  size_t base = (size_t)(b * H_ + h) * WN;
  float o;
  if (p == 0) {
    o = outbuf[(base + 0) * 4096 + (size_t)(2 * n) * 64 + d];
  } else if (p == 63) {
    o = outbuf[(base + 62) * 4096 + (size_t)(2 * n + 1) * 64 + d];
  } else {
    float e1 = outbuf[(base + (p - 1)) * 4096 + (size_t)(2 * n + 1) * 64 + d];
    float e2 = outbuf[(base + p) * 4096 + (size_t)(2 * n) * 64 + d];
    o = 0.5f * (e1 + e2);
  }
  out[idx] = o;
}

// ---------------------------------------------------------------------------
// K6: graph_stable_loss = mean over (b,h,i,j) of var over w
// ---------------------------------------------------------------------------
__global__ void loss_kernel(const float* __restrict__ gsum, const float* __restrict__ gsq,
                            float* __restrict__ lossp) {
  __shared__ float red[256];
  int idx = blockIdx.x * 256 + threadIdx.x;
  float m = gsum[idx] * (1.0f / 63.0f);
  float var = gsq[idx] * (1.0f / 63.0f) - m * m;
  red[threadIdx.x] = var * (1.0f / (float)TOT_STAT);
  __syncthreads();
  for (int s = 128; s > 0; s >>= 1) {
    if (threadIdx.x < s) red[threadIdx.x] += red[threadIdx.x + s];
    __syncthreads();
  }
  if (threadIdx.x == 0) atomicAdd(lossp, red[0]);
}

// ---------------------------------------------------------------------------
extern "C" void kernel_launch(void* const* d_in, const int* in_sizes, int n_in,
                              void* d_out, int out_size, void* d_ws, size_t ws_size,
                              hipStream_t stream) {
  (void)in_sizes; (void)n_in; (void)out_size; (void)ws_size;
  const float* x     = (const float*)d_in[0];
  const float* W1    = (const float*)d_in[1];
  const float* b1    = (const float*)d_in[2];
  const float* W2    = (const float*)d_in[3];
  const float* b2    = (const float*)d_in[4];
  const float* Wv    = (const float*)d_in[5];
  const float* bv    = (const float*)d_in[6];
  const float* alpha = (const float*)d_in[7];

  char* ws = (char*)d_ws;
  __bf16* wxA    = (__bf16*)(ws + OFF_WXA);
  __bf16* vt     = (__bf16*)(ws + OFF_VT);
  float*  outbuf = (float*)(ws + OFF_OUTB);
  __bf16* W1t    = (__bf16*)(ws + OFF_W1T);
  __bf16* W2t    = (__bf16*)(ws + OFF_W2T);
  __bf16* WvT    = (__bf16*)(ws + OFF_WVT);
  float*  gsum   = (float*)(ws + OFF_GSUM);
  float*  gsq    = (float*)(ws + OFF_GSQ);

  float* outF  = (float*)d_out;
  float* lossp = outF + TOT_OUT;

  zero_kernel<<<dim3((unsigned)((2 * TOT_STAT + 255) / 256)), dim3(256), 0, stream>>>(gsum, lossp);
  prep_kernel<<<dim3(288), dim3(256), 0, stream>>>(W1, W2, Wv, W1t, W2t, WvT);
  gather_wx_kernel<<<dim3((unsigned)((TOT_WX + 255) / 256)), dim3(256), 0, stream>>>(x, wxA);
  v_kernel<<<dim3(B_ * WN), dim3(256), 0, stream>>>(wxA, WvT, bv, vt);
  qk_adj_out_kernel<<<dim3(B_ * H_ * WN), dim3(256), 0, stream>>>(
      wxA, vt, W1t, W2t, b1, b2, alpha, gsum, gsq, outbuf);
  merge_kernel<<<dim3((unsigned)((TOT_OUT + 255) / 256)), dim3(256), 0, stream>>>(outbuf, outF);
  loss_kernel<<<dim3((unsigned)(TOT_STAT / 256)), dim3(256), 0, stream>>>(gsum, gsq, lossp);
}
